// SkipTransformer_11364483465522
// MI455X (gfx1250) — matrix operand order repaired and checked by reading, not compiled, not run on hardware
//
#include <hip/hip_runtime.h>
#include <math.h>

// ---------------------------------------------------------------------------
// SkipTransformer for MI455X (gfx1250, wave32).
// ~200 GFLOP of 1x1-conv / attention GEMMs, <50MB HBM traffic -> pure matrix-
// compute bound -> everything matmul-shaped runs on V_WMMA_F32_16X16X32_BF16.
// All GEMM operands are pre-converted to bf16 so inner loops are pure
// b128 loads + WMMA; 4x M register blocking (each B fragment feeds 4 WMMAs)
// and double-buffered fragment loads so waits are partial (loads for k+32 in
// flight while the WMMAs for k execute).
// ---------------------------------------------------------------------------

typedef unsigned int u32;
typedef unsigned short u16;
typedef __attribute__((ext_vector_type(16))) __bf16 v16bf;
typedef __attribute__((ext_vector_type(8)))  float  v8f;

#define NPTS   1024
#define NB     2
#define KNN    20
#define DIM    256
#define HID    1024
#define ROWS2  2048   /* = DIM * UP */
#define UPF    8

// ---- bf16 helpers (RNE) ----------------------------------------------------
__device__ __forceinline__ u16 f2bf(float f) {
    u32 u = __float_as_uint(f);
    return (u16)((u + 0x7FFFu + ((u >> 16) & 1u)) >> 16);
}
__device__ __forceinline__ float bf2f(u16 h) {
    return __uint_as_float(((u32)h) << 16);
}
__device__ __forceinline__ u32 packbf(float a, float b) {
    return (u32)f2bf(a) | ((u32)f2bf(b) << 16);
}

// ---- fragments: two 16B spans per lane -------------------------------------
struct Frag { uint4 lo, hi; };

// A fragment: 16x32 bf16 tile of row-major W (ld halfwords).
// ISA 7.12.2: lane m=l%16, regs j<4 -> K=2j+8*(l/16); j>=4 -> +16.
__device__ __forceinline__ Frag loadA4(const u16* __restrict__ W, int ld,
                                       int m0, int k0) {
    int lane = threadIdx.x & 31;
    int h8 = (lane >> 4) << 3;
    const u16* p = W + (size_t)(m0 + (lane & 15)) * ld + k0 + h8;
    Frag f;
    f.lo = *(const uint4*)p;
    f.hi = *(const uint4*)(p + 16);
    return f;
}

// B fragment: 32x16 bf16 tile from k-contiguous storage base[col][ld].
__device__ __forceinline__ Frag loadB4(const u16* base, int ld,
                                       int n0, int k0, int maxCol) {
    int lane = threadIdx.x & 31;
    int col = n0 + (lane & 15);
    if (col > maxCol) col = maxCol;   // duplicate valid column (dead output)
    const u16* p = base + (size_t)col * ld + k0 + ((lane >> 4) << 3);
    Frag f;
    f.lo = *(const uint4*)p;
    f.hi = *(const uint4*)(p + 16);
    return f;
}

__device__ __forceinline__ v8f wmma_bf16(const Frag& A, const Frag& B, v8f c) {
    union U { uint4 q[2]; v16bf v; } a, b;
    a.q[0] = A.lo; a.q[1] = A.hi;
    b.q[0] = B.lo; b.q[1] = B.hi;
    return __builtin_amdgcn_wmma_f32_16x16x32_bf16(false, a.v, false, b.v,
                                                   (short)0, c, false, false);
}

// 4 row-tiles x 2 col-tiles block GEMM over K=KT with double-buffered
// fragment loads: loads for k+32 are issued before the WMMAs for k, so the
// hardware only needs partial LOADcnt/DScnt waits (latency overlapped).
template <int KT>
__device__ __forceinline__ void mm4x2(const u16* __restrict__ Aw, int ldA, int m0,
                                      const u16* Bs, int ldB,
                                      v8f acc0[4], v8f acc1[4]) {
    Frag cB0 = loadB4(Bs, ldB, 0,  0, KNN - 1);
    Frag cB1 = loadB4(Bs, ldB, 16, 0, KNN - 1);
    Frag cA[4];
#pragma unroll
    for (int r = 0; r < 4; ++r) cA[r] = loadA4(Aw, ldA, m0 + r * 16, 0);
    for (int k0 = 0; k0 < KT - 32; k0 += 32) {
        Frag nB0 = loadB4(Bs, ldB, 0,  k0 + 32, KNN - 1);
        Frag nB1 = loadB4(Bs, ldB, 16, k0 + 32, KNN - 1);
        Frag nA[4];
#pragma unroll
        for (int r = 0; r < 4; ++r) nA[r] = loadA4(Aw, ldA, m0 + r * 16, k0 + 32);
#pragma unroll
        for (int r = 0; r < 4; ++r) {
            acc0[r] = wmma_bf16(cA[r], cB0, acc0[r]);
            acc1[r] = wmma_bf16(cA[r], cB1, acc1[r]);
        }
        cB0 = nB0; cB1 = nB1;
#pragma unroll
        for (int r = 0; r < 4; ++r) cA[r] = nA[r];
    }
#pragma unroll
    for (int r = 0; r < 4; ++r) {
        acc0[r] = wmma_bf16(cA[r], cB0, acc0[r]);
        acc1[r] = wmma_bf16(cA[r], cB1, acc1[r]);
    }
}

// ---------------------------------------------------------------------------
// Weight/activation prep: bf16 conversion + BN folding + aT transpose + kqT.
// ---------------------------------------------------------------------------
__global__ void prep_kernel(
    const float* key, const float* query,
    const float* v1W, const float* v2W, const float* vsW,
    const float* keyW, const float* queryW, const float* valueW,
    const float* pos2W, const float* a1W, const float* aTW, const float* endW,
    const float* ag, const float* abe, const float* am, const float* av, const float* a1b,
    const float* pg, const float* pbe, const float* pm, const float* pv,
    u16* kqT,
    u16* v1bf, u16* v2bf, u16* vsbf,
    u16* keybf, u16* querybf, u16* valuebf,
    u16* pos2bf, u16* a1bf, u16* wtbf, u16* endbf,
    float* a1bias, float* posinv, float* posoff)
{
    int t = blockIdx.x * blockDim.x + threadIdx.x;
    int s = gridDim.x * blockDim.x;
    for (int i = t; i < 128 * 256; i += s) v1bf[i]  = f2bf(v1W[i]);
    for (int i = t; i < 128 * 128; i += s) v2bf[i]  = f2bf(v2W[i]);
    for (int i = t; i < 128 * 256; i += s) vsbf[i]  = f2bf(vsW[i]);
    for (int i = t; i < 256 * 128; i += s) {
        keybf[i]   = f2bf(keyW[i]);
        querybf[i] = f2bf(queryW[i]);
        valuebf[i] = f2bf(valueW[i]);
    }
    for (int i = t; i < 256 * 64;  i += s) pos2bf[i] = f2bf(pos2W[i]);
    for (int i = t; i < 128 * 256; i += s) endbf[i]  = f2bf(endW[i]);
    for (int c = t; c < HID; c += s) {
        float inv = ag[c] / sqrtf(av[c] + 1e-5f);
        a1bias[c] = a1b[c] * inv + abe[c] - am[c] * inv;
    }
    for (int i = t; i < HID * DIM; i += s) {
        int c = i >> 8;
        float inv = ag[c] / sqrtf(av[c] + 1e-5f);
        a1bf[i] = f2bf(a1W[i] * inv);
    }
    for (int i = t; i < 64; i += s) {
        float inv = pg[i] / sqrtf(pv[i] + 1e-5f);
        posinv[i] = inv;
        posoff[i] = pbe[i] - pm[i] * inv;
    }
    // Wt[row=o*8+r][c] = aT_W[c, o, r]
    for (int i = t; i < ROWS2 * HID; i += s) {
        int row = i >> 10, c = i & 1023;
        wtbf[i] = f2bf(aTW[(size_t)c * ROWS2 + row]);
    }
    // kqT[b][n][c] = concat(key,query)[b][c][n]  (bf16, k-contiguous for GEMM B)
    for (int i = t; i < NB * NPTS * 256; i += s) {
        int b = i / (NPTS * 256);
        int r = i - b * NPTS * 256;
        int nn = r >> 8, c = r & 255;
        float v = (c < 128)
            ? key  [(size_t)b * 128 * NPTS + (size_t)c * NPTS + nn]
            : query[(size_t)b * 128 * NPTS + (size_t)(c - 128) * NPTS + nn];
        kqT[i] = f2bf(v);
    }
}

// ---------------------------------------------------------------------------
// Stable KNN (matches jnp.argsort stable tie-break): identical distance
// formula; strict-< insertion keeps the earlier index on ties.
// ---------------------------------------------------------------------------
__global__ void __launch_bounds__(256) knn_kernel(const float* __restrict__ pos,
                                                  int* __restrict__ idxg)
{
    __shared__ float sx[NPTS], sy[NPTS], sz[NPTS], sq[NPTS];
    int tid = threadIdx.x;
    int q = blockIdx.x * 256 + tid;
    int b = q >> 10, n = q & 1023;
    const float* pb = pos + (size_t)b * 3 * NPTS;
    for (int m = tid; m < NPTS; m += 256) {
        float x = pb[m], y = pb[NPTS + m], z = pb[2 * NPTS + m];
        sx[m] = x; sy[m] = y; sz[m] = z; sq[m] = x * x + y * y + z * z;
    }
    __syncthreads();
    float bd[KNN]; int bi[KNN];
#pragma unroll
    for (int j = 0; j < KNN; ++j) { bd[j] = 3.4e38f; bi[j] = 0; }
    float qx = sx[n], qy = sy[n], qz = sz[n], qs = sq[n];
    for (int m = 0; m < NPTS; ++m) {
        float d = qs + sq[m] - 2.f * (qx * sx[m] + qy * sy[m] + qz * sz[m]);
        if (d < bd[KNN - 1]) {
            int p = KNN - 1;
            while (p > 0 && d < bd[p - 1]) { bd[p] = bd[p - 1]; bi[p] = bi[p - 1]; --p; }
            bd[p] = d; bi[p] = m;
        }
    }
    int* op = idxg + (size_t)q * KNN;
    for (int j = 0; j < KNN; ++j) op[j] = bi[j];
}

// ---------------------------------------------------------------------------
// Generic WMMA GEMM: Out = act(W(M,Cin) @ X + bias) [+addC][+addUp].
// X supplied bf16-TRANSPOSED: XT[b][col][k], ld = ldT halfwords.
// grid = (M/64, npts/64, B), block = 128. Wave w: cols (blockIdx.y*4+w)*16,
// rows blockIdx.x*64 .. +63 (4 row-tiles, B fragment reused 4x); fragment
// loads double-buffered. Optionally also writes bf16-transposed OutT.
// ---------------------------------------------------------------------------
__global__ void __launch_bounds__(128) wmma_gemm(
    const u16* __restrict__ Wbf, int M, int Cin,
    const u16* __restrict__ XT, int ldT, int nptsX,
    const float* __restrict__ bias, int doRelu,
    const float* __restrict__ addC,
    const float* __restrict__ addUp, int upDiv, int ldUp,
    float* __restrict__ Out, int ldOut,
    u16* __restrict__ OutT, int ldOT)
{
    int b = blockIdx.z;
    int wave = threadIdx.x >> 5;
    int lane = threadIdx.x & 31;
    int col0 = (blockIdx.y * 4 + wave) * 16;
    int mg = blockIdx.x * 64;
    const u16* XTb = XT + (size_t)b * nptsX * ldT;

    v8f acc[4] = {};
    Frag cB = loadB4(XTb, ldT, col0, 0, 0x7fffffff);
    Frag cA[4];
#pragma unroll
    for (int r = 0; r < 4; ++r) cA[r] = loadA4(Wbf, Cin, mg + r * 16, 0);
    for (int k0 = 0; k0 < Cin - 32; k0 += 32) {
        Frag nB = loadB4(XTb, ldT, col0, k0 + 32, 0x7fffffff);
        Frag nA[4];
#pragma unroll
        for (int r = 0; r < 4; ++r) nA[r] = loadA4(Wbf, Cin, mg + r * 16, k0 + 32);
#pragma unroll
        for (int r = 0; r < 4; ++r) acc[r] = wmma_bf16(cA[r], cB, acc[r]);
        cB = nB;
#pragma unroll
        for (int r = 0; r < 4; ++r) cA[r] = nA[r];
    }
#pragma unroll
    for (int r = 0; r < 4; ++r) acc[r] = wmma_bf16(cA[r], cB, acc[r]);

    int col = col0 + (lane & 15);
    size_t outB = (size_t)b * M * ldOut;
#pragma unroll
    for (int r = 0; r < 4; ++r) {
        int rbase = mg + r * 16 + ((lane >> 4) << 3);
        float vals[8];
#pragma unroll
        for (int v = 0; v < 8; ++v) {
            int row = rbase + v;
            float val = acc[r][v] + bias[row];
            if (doRelu) val = fmaxf(val, 0.f);
            if (addC)  val += addC[outB + (size_t)row * ldOut + col];
            if (addUp) val += addUp[(size_t)b * M * ldUp + (size_t)row * ldUp + col / upDiv];
            vals[v] = val;
        }
        if (Out) {
#pragma unroll
            for (int v = 0; v < 8; ++v)
                Out[outB + (size_t)(rbase + v) * ldOut + col] = vals[v];
        }
        if (OutT) {
            uint4 pk;
            pk.x = packbf(vals[0], vals[1]);
            pk.y = packbf(vals[2], vals[3]);
            pk.z = packbf(vals[4], vals[5]);
            pk.w = packbf(vals[6], vals[7]);
            *(uint4*)(OutT + (size_t)b * nptsX * ldOT + (size_t)col * ldOT + rbase) = pk;
        }
    }
}

// ---------------------------------------------------------------------------
// Fused per-point attention. 1 workgroup = 1 (b,n) point, 8 waves.
//   pe  = pos2 @ relu(bn(pos1 @ pos_rel))          (WMMA, 256x20)
//   X   = q_f - k_g + pe                           (bf16 LDS, 256x20)
//   h   = relu(a1s @ X + a1bias)                   (WMMA 4x-blocked, 40KB LDS)
//   att = Wt(2048,1024) @ h  (WMMA 4x-blocked, pipelined); softmax over the
//         20 neighbors done inside the C-fragments (16-lane shfl reductions);
//         weighted v_g sum written straight to aggT (bf16 transposed).
// LDS ~ 64KB -> 2+ workgroups per 320KB WGP.
// ---------------------------------------------------------------------------
__global__ void __launch_bounds__(256) fused_attn(
    const float* __restrict__ pos,
    const float* __restrict__ qf, const float* __restrict__ kf,
    const float* __restrict__ vf,
    const int* __restrict__ idxg,
    const float* __restrict__ pos1W, const float* __restrict__ pos1b,
    const float* __restrict__ posinv, const float* __restrict__ posoff,
    const u16* __restrict__ pos2bf, const float* __restrict__ pos2b,
    const u16* __restrict__ a1bf, const float* __restrict__ a1bias,
    const u16* __restrict__ wtbf, const float* __restrict__ aTb,
    u16* __restrict__ aggT)
{
    __shared__ int sIdx[KNN];
    __shared__ u16 sT[KNN][64];     // t64, [j][c]
    __shared__ u16 sX[KNN][DIM];    // X,   [j][c]
    __shared__ u16 sH[KNN][HID];    // h,   [j][c]
    __shared__ u16 sV[DIM][KNN];    // pe, then v_g (bf16)

    int tid = threadIdx.x;
    int wave = tid >> 5;
    int lane = tid & 31;
    int ln = lane & 15, half = lane >> 4;
    int n = blockIdx.x, bI = blockIdx.y;

    if (tid < KNN) sIdx[tid] = idxg[((size_t)bI * NPTS + n) * KNN + tid];
    __syncthreads();

    // (a) t64 = relu(bn(pos1 @ pos_rel))  (64x20, VALU — <1% of FLOPs)
    for (int e = tid; e < 64 * KNN; e += 256) {
        int c = e / KNN, j = e % KNN;
        int nb = sIdx[j];
        const float* pb = pos + (size_t)bI * 3 * NPTS;
        float acc = pos1b[c];
#pragma unroll
        for (int d = 0; d < 3; ++d)
            acc += pos1W[c * 3 + d] * (pb[d * NPTS + n] - pb[d * NPTS + nb]);
        sT[j][c] = f2bf(fmaxf(acc * posinv[c] + posoff[c], 0.f));
    }
    __syncthreads();

    // (b) pe = pos2 @ t64 + pos2_b  (WMMA, 16 row-tiles x 2 col-tiles)
    for (int t = wave; t < 32; t += 8) {
        int rt = t >> 1, ct = t & 1;
        v8f acc = {};
#pragma unroll
        for (int k0 = 0; k0 < 64; k0 += 32) {
            Frag Af = loadA4(pos2bf, 64, rt * 16, k0);
            Frag Bf = loadB4(&sT[0][0], 64, ct * 16, k0, KNN - 1);
            acc = wmma_bf16(Af, Bf, acc);
        }
        int j = ct * 16 + ln;
        if (j < KNN) {
            int rbase = rt * 16 + half * 8;
#pragma unroll
            for (int v = 0; v < 8; ++v)
                sV[rbase + v][j] = f2bf(acc[v] + pos2b[rbase + v]);
        }
    }
    __syncthreads();

    // (c) X = q_f - k_g + pe
    for (int e = tid; e < DIM * KNN; e += 256) {
        int c = e / KNN, j = e % KNN;
        size_t base = (size_t)bI * DIM * NPTS + (size_t)c * NPTS;
        float val = qf[base + n] - kf[base + sIdx[j]] + bf2f(sV[c][j]);
        sX[j][c] = f2bf(val);
    }
    __syncthreads();

    // (d) h = relu(a1s @ X + a1bias)  — 64 row-tiles, 4x M-blocked, pipelined
    for (int g = 0; g < 2; ++g) {
        int rtb = wave * 8 + g * 4;
        v8f acc0[4] = {}, acc1[4] = {};
        mm4x2<DIM>(a1bf, DIM, rtb * 16, &sX[0][0], DIM, acc0, acc1);
#pragma unroll
        for (int r = 0; r < 4; ++r) {
            int rbase = (rtb + r) * 16 + half * 8;
#pragma unroll
            for (int v = 0; v < 8; ++v) {
                int row = rbase + v;
                sH[ln][row] = f2bf(fmaxf(acc0[r][v] + a1bias[row], 0.f));
                if (ln < KNN - 16)
                    sH[16 + ln][row] = f2bf(fmaxf(acc1[r][v] + a1bias[row], 0.f));
            }
        }
    }
    // (e) v_g = v_f_gathered + pe, in place over sV (own-element RMW)
    for (int e = tid; e < DIM * KNN; e += 256) {
        int c = e / KNN, j = e % KNN;
        size_t base = (size_t)bI * DIM * NPTS + (size_t)c * NPTS;
        sV[c][j] = f2bf(vf[base + sIdx[j]] + bf2f(sV[c][j]));
    }
    __syncthreads();

    // (f) att = Wt @ h ; softmax + weighted v_g sum in-fragment.
    for (int g = 0; g < 4; ++g) {
        int rtb = wave * 16 + g * 4;
        v8f acc0[4] = {}, acc1[4] = {};
        mm4x2<HID>(wtbf, HID, rtb * 16, &sH[0][0], HID, acc0, acc1);
#pragma unroll
        for (int r = 0; r < 4; ++r) {
            int rt = rtb + r;
#pragma unroll
            for (int v = 0; v < 8; ++v) {
                int row = rt * 16 + half * 8 + v;   // row = o*8 + up
                int o = row >> 3;
                float bsc = aTb[o];
                float t0 = acc0[r][v] + bsc;                          // j = ln
                float t1 = (ln < 4) ? (acc1[r][v] + bsc) : -3.4e38f;  // j = 16+ln
                float m = fmaxf(t0, t1);
#pragma unroll
                for (int d = 1; d < 16; d <<= 1) m = fmaxf(m, __shfl_xor(m, d, 16));
                float e0 = __expf(t0 - m);
                float e1 = (ln < 4) ? __expf(t1 - m) : 0.f;
                float den = e0 + e1;
                float num = e0 * bf2f(sV[o][ln]);
                if (ln < 4) num += e1 * bf2f(sV[o][16 + ln]);
#pragma unroll
                for (int d = 1; d < 16; d <<= 1) {
                    den += __shfl_xor(den, d, 16);
                    num += __shfl_xor(num, d, 16);
                }
                if (ln == 0)   // aggT[b][n*8+up][o]  (bf16 transposed, ld=256)
                    aggT[(size_t)bI * NPTS * UPF * DIM
                         + ((size_t)n * UPF + (row & 7)) * DIM + o] = f2bf(num / den);
            }
        }
    }
}

// ---------------------------------------------------------------------------
extern "C" void kernel_launch(void* const* d_in, const int* in_sizes, int n_in,
                              void* d_out, int out_size, void* d_ws, size_t ws_size,
                              hipStream_t stream)
{
    const float* pos     = (const float*)d_in[0];
    const float* key     = (const float*)d_in[1];
    const float* query   = (const float*)d_in[2];
    const float* v1_W    = (const float*)d_in[3];
    const float* v1_b    = (const float*)d_in[4];
    const float* v2_W    = (const float*)d_in[5];
    const float* v2_b    = (const float*)d_in[6];
    const float* vs_W    = (const float*)d_in[7];
    const float* vs_b    = (const float*)d_in[8];
    const float* key_W   = (const float*)d_in[9];
    const float* key_b   = (const float*)d_in[10];
    const float* query_W = (const float*)d_in[11];
    const float* query_b = (const float*)d_in[12];
    const float* value_W = (const float*)d_in[13];
    const float* value_b = (const float*)d_in[14];
    const float* pos1_W  = (const float*)d_in[15];
    const float* pos1_b  = (const float*)d_in[16];
    const float* pos_g   = (const float*)d_in[17];
    const float* pos_be  = (const float*)d_in[18];
    const float* pos_m   = (const float*)d_in[19];
    const float* pos_v   = (const float*)d_in[20];
    const float* pos2_W  = (const float*)d_in[21];
    const float* pos2_b  = (const float*)d_in[22];
    const float* a1_W    = (const float*)d_in[23];
    const float* a1_b    = (const float*)d_in[24];
    const float* a_g     = (const float*)d_in[25];
    const float* a_be    = (const float*)d_in[26];
    const float* a_m     = (const float*)d_in[27];
    const float* a_v     = (const float*)d_in[28];
    const float* aT_W    = (const float*)d_in[29];
    const float* aT_b    = (const float*)d_in[30];
    const float* end_W   = (const float*)d_in[31];
    const float* end_b   = (const float*)d_in[32];

    // ---- workspace carve (~23 MB) ----
    char* p = (char*)d_ws;
    auto take = [&](size_t bytes) -> void* {
        void* r = (void*)p;
        p += (bytes + 255) & ~(size_t)255;
        return r;
    };
    int*   idxw    = (int*)  take((size_t)NB * NPTS * KNN * 4);
    float* value   = (float*)take((size_t)NB * 128 * NPTS * 4);
    float* kfb     = (float*)take((size_t)NB * DIM * NPTS * 4);
    float* qfb     = (float*)take((size_t)NB * DIM * NPTS * 4);
    float* vfb     = (float*)take((size_t)NB * DIM * NPTS * 4);
    u16* kqT       = (u16*)take((size_t)NB * NPTS * 256 * 2);
    u16* tmp1T     = (u16*)take((size_t)NB * NPTS * 128 * 2);
    u16* valueT    = (u16*)take((size_t)NB * NPTS * 128 * 2);
    u16* aggT      = (u16*)take((size_t)NB * NPTS * UPF * DIM * 2);
    u16* v1bf      = (u16*)take(128 * 256 * 2);
    u16* v2bf      = (u16*)take(128 * 128 * 2);
    u16* vsbf      = (u16*)take(128 * 256 * 2);
    u16* keybf     = (u16*)take(256 * 128 * 2);
    u16* querybf   = (u16*)take(256 * 128 * 2);
    u16* valuebf   = (u16*)take(256 * 128 * 2);
    u16* pos2bf    = (u16*)take(256 * 64 * 2);
    u16* a1bf      = (u16*)take((size_t)HID * DIM * 2);
    u16* wtbf      = (u16*)take((size_t)ROWS2 * HID * 2);
    u16* endbf     = (u16*)take(128 * 256 * 2);
    float* a1bias  = (float*)take(HID * 4);
    float* posinv  = (float*)take(64 * 4);
    float* posoff  = (float*)take(64 * 4);

    // 1) prep: bf16 weights, BN folds, aT transpose, kqT activations
    prep_kernel<<<256, 256, 0, stream>>>(
        key, query,
        v1_W, v2_W, vs_W, key_W, query_W, value_W, pos2_W, a1_W, aT_W, end_W,
        a_g, a_be, a_m, a_v, a1_b, pos_g, pos_be, pos_m, pos_v,
        kqT, v1bf, v2bf, vsbf, keybf, querybf, valuebf, pos2bf, a1bf, wtbf, endbf,
        a1bias, posinv, posoff);

    // 2) KNN indices
    knn_kernel<<<NB * NPTS / 256, 256, 0, stream>>>(pos, idxw);

    // 3) feature GEMMs (all WMMA; bf16-transposed B operands)
    // tmp1T = relu(v1_W @ [key;query] + v1_b)        (bf16-T only)
    wmma_gemm<<<dim3(2, 16, NB), 128, 0, stream>>>(
        v1bf, 128, 256, kqT, 256, NPTS, v1_b, 1,
        nullptr, nullptr, 1, NPTS, nullptr, NPTS, tmp1T, 128);
    // value = vs_W @ [key;query] + vs_b              (f32)
    wmma_gemm<<<dim3(2, 16, NB), 128, 0, stream>>>(
        vsbf, 128, 256, kqT, 256, NPTS, vs_b, 0,
        nullptr, nullptr, 1, NPTS, value, NPTS, nullptr, 128);
    // value += v2_W @ tmp1 + v2_b                    (f32 + bf16-T)
    wmma_gemm<<<dim3(2, 16, NB), 128, 0, stream>>>(
        v2bf, 128, 128, tmp1T, 128, NPTS, v2_b, 0,
        value, nullptr, 1, NPTS, value, NPTS, valueT, 128);
    // k_f / q_f / v_f  (f32 for fused gather path)
    wmma_gemm<<<dim3(4, 16, NB), 128, 0, stream>>>(
        keybf, 256, 128, kqT, 256, NPTS, key_b, 0,
        nullptr, nullptr, 1, NPTS, kfb, NPTS, nullptr, 128);
    wmma_gemm<<<dim3(4, 16, NB), 128, 0, stream>>>(
        querybf, 256, 128, kqT + 128, 256, NPTS, query_b, 0,
        nullptr, nullptr, 1, NPTS, qfb, NPTS, nullptr, 128);
    wmma_gemm<<<dim3(4, 16, NB), 128, 0, stream>>>(
        valuebf, 256, 128, valueT, 128, NPTS, value_b, 0,
        nullptr, nullptr, 1, NPTS, vfb, NPTS, nullptr, 128);

    // 4) fused attention -> aggT (B, N*8, 256) bf16-T
    fused_attn<<<dim3(NPTS, NB), 256, 0, stream>>>(
        pos, qfb, kfb, vfb, idxw, pos1_W, pos1_b, posinv, posoff,
        pos2bf, pos2_b, a1bf, a1bias, wtbf, aT_b, aggT);

    // 5) out = end_W @ agg + end_b + repeat(value, 8)
    wmma_gemm<<<dim3(2, NPTS * UPF / 64, NB), 128, 0, stream>>>(
        endbf, 128, 256, aggT, 256, NPTS * UPF, end_b, 0,
        nullptr, value, UPF, NPTS, (float*)d_out, NPTS * UPF, nullptr, 256);
}
